// MultiHeadSelfAttentionLrw_28544352650045
// MI455X (gfx1250) — matrix operand order repaired and checked by reading, not compiled, not run on hardware
//
#include <hip/hip_runtime.h>
#include <hip/hip_bf16.h>

typedef __attribute__((ext_vector_type(16))) _Float16 v16h;
typedef __attribute__((ext_vector_type(8)))  _Float16 v8h;
typedef __attribute__((ext_vector_type(2)))  _Float16 v2h;
typedef __attribute__((ext_vector_type(8)))  float    v8f;

#define L_SEQ 2048
#define NBATCH 2
#define EDIM 1024
#define HEADS 16
#define INNER_DIM 1024
#define DHEAD 64
#define NHB (NBATCH * HEADS)      // 32 attention batches
#define ROWS (L_SEQ * NBATCH)     // 4096 GEMM rows
#define QK_SCALE 0.125f           // D^-0.5 = 64^-0.5

static __device__ __forceinline__ v16h pack16(v8h lo, v8h hi) {
    v16h r;
#pragma unroll
    for (int i = 0; i < 8; ++i) { r[i] = lo[i]; r[8 + i] = hi[i]; }
    return r;
}

static __device__ __forceinline__ v8f wmma_f16(v16h a, v16h b, v8f c) {
    // D = A(16x32 f16) * B(32x16 f16) + C(16x16 f32)
    return __builtin_amdgcn_wmma_f32_16x16x32_f16(
        /*neg_a=*/false, a, /*neg_b=*/false, b,
        /*c_mod=*/(short)0, c, /*reuse_a=*/false, /*reuse_b=*/false);
}

// CDNA5 async copy: 16 bytes global -> LDS, tracked by ASYNCcnt.
// LDS operand is the low-32-bit byte offset of the generic shared pointer.
static __device__ __forceinline__ void async_copy16(void* lds_ptr,
                                                    const void* gptr) {
    unsigned int lds_off = (unsigned int)(size_t)lds_ptr;
    asm volatile("global_load_async_to_lds_b128 %0, %1, off"
                 :: "v"(lds_off), "v"(gptr)
                 : "memory");
}
static __device__ __forceinline__ void wait_async0() {
    asm volatile("s_wait_asynccnt 0x0" ::: "memory");
}
static __device__ __forceinline__ void wait_async2() {
    asm volatile("s_wait_asynccnt 0x2" ::: "memory");
}
static __device__ __forceinline__ void wait_async4() {
    asm volatile("s_wait_asynccnt 0x4" ::: "memory");
}

// max-reduce across each 16-lane group: VALU permlane16 butterflies
// (keeps the LDS pipe free for WMMA fragment traffic); shuffle fallback.
static __device__ __forceinline__ float rowmax16(float v) {
#if __has_builtin(__builtin_amdgcn_permlane16)
    unsigned u, p;
    u = __float_as_uint(v);   // xor 1
    p = __builtin_amdgcn_permlane16(u, u, 0x67452301u, 0xEFCDAB89u, false, false);
    v = fmaxf(v, __uint_as_float(p));
    u = __float_as_uint(v);   // xor 2
    p = __builtin_amdgcn_permlane16(u, u, 0x45670123u, 0xCDEF89ABu, false, false);
    v = fmaxf(v, __uint_as_float(p));
    u = __float_as_uint(v);   // xor 4
    p = __builtin_amdgcn_permlane16(u, u, 0x01234567u, 0x89ABCDEFu, false, false);
    v = fmaxf(v, __uint_as_float(p));
    u = __float_as_uint(v);   // xor 8
    p = __builtin_amdgcn_permlane16(u, u, 0xFEDCBA98u, 0x76543210u, false, false);
    v = fmaxf(v, __uint_as_float(p));
    return v;
#else
#pragma unroll
    for (int off = 8; off; off >>= 1) v = fmaxf(v, __shfl_xor(v, off, 32));
    return v;
#endif
}

// ---------------------------------------------------------------------------
// f32 -> f16 convert (one pass so all GEMM staging becomes raw async copies)
// ---------------------------------------------------------------------------
__global__ __launch_bounds__(256) void cvt_f32_to_f16(
    const float* __restrict__ s, _Float16* __restrict__ d, int n)
{
    int i = (blockIdx.x * blockDim.x + threadIdx.x) * 8;
    if (i < n) {
#pragma unroll
        for (int j = 0; j < 8; ++j) d[i + j] = (_Float16)s[i + j];
    }
}

// ---------------------------------------------------------------------------
// GEMM: C[R][F] = sum_e A[R][e] * B[F][e] + bias[F]   (A,B f16 row-major,
// contraction over columns of both). Block: 256 threads (8 waves).
// Tile 128(M) x 128(N), K step 32, double-buffered async-to-LDS staging;
// each wave computes 16x128 (8 WMMAs / K-step reusing one A fragment).
//   EPI==0: epilogue de-interleaves stride-3 q/k/v, scatters f16 Qb/Kb/Vb
//           laid out [NHB][L][D].
//   EPI==1: epilogue writes f32 result + bias to Co.
// ---------------------------------------------------------------------------
template <int EPI>
__global__ __launch_bounds__(256) void gemm_kernel(
    const _Float16* __restrict__ A, const _Float16* __restrict__ B,
    const float* __restrict__ bias,
    _Float16* __restrict__ Qb, _Float16* __restrict__ Kb,
    _Float16* __restrict__ Vb, float* __restrict__ Co)
{
    constexpr int K  = 1024;
    constexpr int NT = K / 32;
    __shared__ _Float16 As[2][128][32];
    __shared__ _Float16 Bs[2][128][32];

    const int tid  = threadIdx.x;
    const int lane = tid & 31;
    const int wv   = tid >> 5;                 // wave 0..7 -> 16 M-rows each
    const int rowBase = blockIdx.y * 128;
    const int colBase = blockIdx.x * 128;

    v8f acc[8];
#pragma unroll
    for (int t = 0; t < 8; ++t)
#pragma unroll
        for (int j = 0; j < 8; ++j) acc[t][j] = 0.0f;

    const int m    = lane & 15;
    const int klo  = (lane < 16) ? 0 : 8;
    const int koff = (lane < 16) ? 0 : 16;

    // A and B each: 512 16B chunks, 2 per thread
    auto issue_tile = [&](int k0, int buf) {
#pragma unroll
        for (int j = 0; j < 2; ++j) {
            const int c = tid * 2 + j;
            const int r = c >> 2, part = (c & 3) * 8;
            async_copy16(&As[buf][r][part],
                         &A[(size_t)(rowBase + r) * K + k0 + part]);
            async_copy16(&Bs[buf][r][part],
                         &B[(size_t)(colBase + r) * K + k0 + part]);
        }
    };

    issue_tile(0, 0);
    for (int it = 0; it < NT; ++it) {
        if (it + 1 < NT) {
            issue_tile((it + 1) * 32, (it + 1) & 1);
            wait_async4();   // tile `it` complete; tile `it+1` still in flight
        } else {
            wait_async0();
        }
        __syncthreads();

        const int buf = it & 1;
        v16h afrag = pack16(*(const v8h*)&As[buf][wv * 16 + m][koff],
                            *(const v8h*)&As[buf][wv * 16 + m][koff + 8]);
#pragma unroll
        for (int t = 0; t < 8; ++t) {
            v16h bfrag = *(const v16h*)&Bs[buf][t * 16 + (lane & 15)][koff];
            acc[t] = wmma_f16(afrag, bfrag, acc[t]);
        }
        __syncthreads();     // buffer free before it is re-issued
    }

    // epilogue: D-frag layout = lane holds col n, rows r (lanes 0-15) / 8+r
    const int n    = lane & 15;
    const int moff = (lane < 16) ? 0 : 8;
#pragma unroll
    for (int t = 0; t < 8; ++t) {
        const int F = colBase + t * 16 + n;
        const float bval = bias[F];
#pragma unroll
        for (int r = 0; r < 8; ++r) {
            const int R = rowBase + wv * 16 + moff + r;
            float val = acc[t][r] + bval;
            if constexpr (EPI == 0) {
                // F = 3*i + s ; i = h*64 + d ; batch = (R&1)*16 + h ; l = R>>1
                const int s = F % 3;
                const int i = F / 3;
                const int b = (R & 1) * HEADS + (i >> 6);
                const int d = i & 63;
                const int l = R >> 1;
                const size_t idx = ((size_t)b * L_SEQ + l) * DHEAD + d;
                const _Float16 hv = (_Float16)val;
                if (s == 0)      Qb[idx] = hv;
                else if (s == 1) Kb[idx] = hv;
                else             Vb[idx] = hv;
            } else {
                Co[(size_t)R * EDIM + F] = val;
            }
        }
    }
}

// ---------------------------------------------------------------------------
// Flash attention: one block per (batch b, 64 query rows); 4 waves, each wave
// owns 16 query rows and full D=64 in registers. Keys in tiles of 32:
//   - K tile: async-to-LDS, double-buffered (copy overlaps WMMA)
//   - V tile: global loads issued one iteration ahead, transposed store
//     deferred to time of use (single LDS buffer, latency hidden)
//   - online softmax: permlane16 row max, row sums via WMMA (P @ ones)
// ---------------------------------------------------------------------------
__global__ __launch_bounds__(128) void flash_attn_kernel(
    const _Float16* __restrict__ Qb, const _Float16* __restrict__ Kb,
    const _Float16* __restrict__ Vb, _Float16* __restrict__ Ob)
{
    constexpr int NT = L_SEQ / 32;
    __shared__ _Float16 Klds[2][32][64];  // K tiles, row-major [key][d]
    __shared__ _Float16 Vt[64][32];       // V transposed: [d][key]
    __shared__ _Float16 Pl[4][16][32];    // per-wave P bounce (D->A relayout)

    const int tid  = threadIdx.x;
    const int lane = tid & 31;
    const int wv   = tid >> 5;                      // 0..3
    const int b    = blockIdx.y;                    // 0..31
    const int qBase = blockIdx.x * 64 + wv * 16;

    const _Float16* Kbb = Kb + (size_t)b * L_SEQ * DHEAD;
    const _Float16* Vbb = Vb + (size_t)b * L_SEQ * DHEAD;

    const int m    = lane & 15;
    const int klo  = (lane < 16) ? 0 : 8;
    const int koff = (lane < 16) ? 0 : 16;
    const int moff = (lane < 16) ? 0 : 8;

    // Q A-fragments, pre-scaled by D^-0.5 (exact in f16)
    v16h aq[2];
#pragma unroll
    for (int ds = 0; ds < 2; ++ds) {
        const _Float16* qp =
            Qb + ((size_t)b * L_SEQ + qBase + m) * DHEAD + ds * 32;
        v8h lo = *(const v8h*)(qp + koff);
        v8h hi = *(const v8h*)(qp + koff + 8);
#pragma unroll
        for (int i = 0; i < 8; ++i) {
            lo[i] = lo[i] * (_Float16)QK_SCALE;
            hi[i] = hi[i] * (_Float16)QK_SCALE;
        }
        aq[ds] = pack16(lo, hi);
    }

    // all-ones B fragment: rowsum(P) = P @ ones via the matrix unit
    v16h ones;
#pragma unroll
    for (int i = 0; i < 16; ++i) ones[i] = (_Float16)1.0f;

    v8f o[4];
    float mrun[8], lrun[8];
#pragma unroll
    for (int t = 0; t < 4; ++t)
#pragma unroll
        for (int j = 0; j < 8; ++j) o[t][j] = 0.0f;
#pragma unroll
    for (int r = 0; r < 8; ++r) { mrun[r] = -3.0e38f; lrun[r] = 0.0f; }

    // V staging: each thread owns a key pair x 8 d values (packed b32 stores)
    const int kpair = (tid >> 3) * 2;     // 0,2,..,30
    const int dgrp  = (tid & 7) * 8;      // 0..56

    auto issue_K = [&](int kb, int buf) {
#pragma unroll
        for (int j = 0; j < 2; ++j) {
            const int c = tid * 2 + j;
            const int key = c >> 3, part = (c & 7) * 8;
            async_copy16(&Klds[buf][key][part],
                         Kbb + (size_t)(kb + key) * DHEAD + part);
        }
    };

    // prologue: V loads + K async copies for tile 0
    v8h va = *(const v8h*)(Vbb + (size_t)kpair * DHEAD + dgrp);
    v8h vb = *(const v8h*)(Vbb + (size_t)(kpair + 1) * DHEAD + dgrp);
    issue_K(0, 0);

    for (int it = 0; it < NT; ++it) {
        const int buf = it & 1;
        // transpose-store this tile's V (loaded last iteration)
#pragma unroll
        for (int i = 0; i < 8; ++i) {
            v2h pr; pr[0] = va[i]; pr[1] = vb[i];
            *(v2h*)&Vt[dgrp + i][kpair] = pr;
        }
        // prefetch next tile: V global loads + K async copies
        if (it + 1 < NT) {
            const int kn = (it + 1) * 32;
            va = *(const v8h*)(Vbb + (size_t)(kn + kpair) * DHEAD + dgrp);
            vb = *(const v8h*)(Vbb + (size_t)(kn + kpair + 1) * DHEAD + dgrp);
            issue_K(kn, (it + 1) & 1);
            wait_async2();   // this tile's K done; next tile's in flight
        } else {
            wait_async0();
        }
        __syncthreads();

        // S = Qs @ K^T for 32 keys: 2 N-subtiles x 2 K-steps over d
        v8f s0, s1;
#pragma unroll
        for (int j = 0; j < 8; ++j) { s0[j] = 0.0f; s1[j] = 0.0f; }
#pragma unroll
        for (int ds = 0; ds < 2; ++ds) {
            v16h bk0 = *(const v16h*)&Klds[buf][lane & 15][ds * 32 + koff];
            v16h bk1 =
                *(const v16h*)&Klds[buf][16 + (lane & 15)][ds * 32 + koff];
            s0 = wmma_f16(aq[ds], bk0, s0);
            s1 = wmma_f16(aq[ds], bk1, s1);
        }

        // online softmax: row max via permlane16 butterflies
        float p0[8], p1[8], alpha[8];
#pragma unroll
        for (int r = 0; r < 8; ++r) {
            const float mx = rowmax16(fmaxf(s0[r], s1[r]));
            const float mn = fmaxf(mrun[r], mx);
            alpha[r] = __expf(mrun[r] - mn);
            p0[r] = __expf(s0[r] - mn);
            p1[r] = __expf(s1[r] - mn);
            mrun[r] = mn;
#pragma unroll
            for (int t = 0; t < 4; ++t) o[t][r] *= alpha[r];
        }

        // P: D-frag layout -> A-frag layout via per-wave LDS bounce
        {
            const int n = lane & 15;
#pragma unroll
            for (int r = 0; r < 8; ++r) {
                Pl[wv][moff + r][n]      = (_Float16)p0[r];
                Pl[wv][moff + r][16 + n] = (_Float16)p1[r];
            }
        }
        v16h ap = pack16(*(const v8h*)&Pl[wv][m][koff],
                         *(const v8h*)&Pl[wv][m][koff + 8]);

        // row sums of (f16-rounded) P on the matrix unit: P @ ones
        {
            v8f zsum;
#pragma unroll
            for (int j = 0; j < 8; ++j) zsum[j] = 0.0f;
            zsum = wmma_f16(ap, ones, zsum);
#pragma unroll
            for (int r = 0; r < 8; ++r)
                lrun[r] = lrun[r] * alpha[r] + zsum[r];
        }

        // O += P @ V (K dim = 32 keys, 4 d-subtiles)
#pragma unroll
        for (int t = 0; t < 4; ++t) {
            v16h bv = *(const v16h*)&Vt[t * 16 + (lane & 15)][koff];
            o[t] = wmma_f16(ap, bv, o[t]);
        }
        __syncthreads();  // Vt/Klds[buf] free before next iteration's stores
    }

    // finalize: divide by row sum, store f16 [L][N][h*64+d]
    {
        const int n  = lane & 15;
        const int nb = b >> 4;
        const int h  = b & 15;
#pragma unroll
        for (int t = 0; t < 4; ++t) {
#pragma unroll
            for (int r = 0; r < 8; ++r) {
                const int lq = qBase + moff + r;
                const float val = o[t][r] / lrun[r];
                Ob[((size_t)lq * NBATCH + nb) * INNER_DIM + h * DHEAD +
                   t * 16 + n] = (_Float16)val;
            }
        }
    }
}

// ---------------------------------------------------------------------------
extern "C" void kernel_launch(void* const* d_in, const int* in_sizes, int n_in,
                              void* d_out, int out_size, void* d_ws,
                              size_t ws_size, hipStream_t stream)
{
    (void)in_sizes; (void)n_in; (void)out_size; (void)ws_size;

    const float* query    = (const float*)d_in[0];  // [L,N,E]
    const float* qkv_proj = (const float*)d_in[1];  // [3*INNER,E]
    const float* qkv_bias = (const float*)d_in[2];  // [3*INNER]
    const float* out_proj = (const float*)d_in[3];  // [E,INNER]
    const float* out_bias = (const float*)d_in[4];  // [E]
    float* out = (float*)d_out;                     // [L,N,E] f32

    // workspace layout (f16)
    _Float16* p  = (_Float16*)d_ws;
    _Float16* qh    = p; p += (size_t)ROWS * EDIM;          // query f16
    _Float16* wqkvh = p; p += (size_t)3 * INNER_DIM * EDIM; // qkv_proj f16
    _Float16* wouth = p; p += (size_t)EDIM * INNER_DIM;     // out_proj f16
    _Float16* Qb    = p; p += (size_t)NHB * L_SEQ * DHEAD;
    _Float16* Kb    = p; p += (size_t)NHB * L_SEQ * DHEAD;
    _Float16* Vb    = p; p += (size_t)NHB * L_SEQ * DHEAD;
    _Float16* Ob    = p; p += (size_t)ROWS * INNER_DIM;

    // 0) one-pass f32 -> f16 conversions (so GEMM staging is raw async copy)
    {
        const int n0 = ROWS * EDIM;
        cvt_f32_to_f16<<<n0 / 8 / 256, 256, 0, stream>>>(query, qh, n0);
        const int n1 = 3 * INNER_DIM * EDIM;
        cvt_f32_to_f16<<<n1 / 8 / 256, 256, 0, stream>>>(qkv_proj, wqkvh, n1);
        const int n2 = EDIM * INNER_DIM;
        cvt_f32_to_f16<<<n2 / 8 / 256, 256, 0, stream>>>(out_proj, wouth, n2);
    }
    // 1) QKV projection + bias + de-interleave scatter (f16 out)
    {
        dim3 grid(3072 / 128, ROWS / 128);
        gemm_kernel<0><<<grid, 256, 0, stream>>>(qh, wqkvh, qkv_bias, Qb, Kb,
                                                 Vb, nullptr);
    }
    // 2) flash attention per (batch, 64-row query tile)
    {
        dim3 grid(L_SEQ / 64, NHB);
        flash_attn_kernel<<<grid, 128, 0, stream>>>(Qb, Kb, Vb, Ob);
    }
    // 3) output projection + bias (f32 out)
    {
        dim3 grid(EDIM / 128, ROWS / 128);
        gemm_kernel<1><<<grid, 256, 0, stream>>>(Ob, wouth, out_bias, nullptr,
                                                 nullptr, nullptr, out);
    }
}
